// HAEModel_56530359549981
// MI455X (gfx1250) — matrix operand order repaired and checked by reading, not compiled, not run on hardware
//
#include <hip/hip_runtime.h>

#define N_NODES 500000
#define N_EDGES 8000000
#define FEAT 10
#define HID 32
#define N_TOOLS 13
#define N_PRIM 8

typedef __attribute__((ext_vector_type(16))) _Float16 v16h;
typedef __attribute__((ext_vector_type(8)))  float    v8f;

// ---- workspace layout (floats) ----
// [0, N_NODES*FEAT)                       agg      (scatter-sum of neighbor feats)
// [N_NODES*FEAT, N_NODES*FEAT+N_NODES)   count
// [+N_NODES, +N_NODES+32)                gsum     (column sums of h for graph embed)

__global__ void zero_ws_kernel(float* __restrict__ ws, int n) {
  int i = blockIdx.x * blockDim.x + threadIdx.x;
  int stride = gridDim.x * blockDim.x;
  for (; i < n; i += stride) ws[i] = 0.0f;
}

__global__ void edge_scatter_kernel(const float* __restrict__ x,
                                    const long long* __restrict__ ei,
                                    float* __restrict__ agg,
                                    float* __restrict__ cnt) {
  int e = blockIdx.x * blockDim.x + threadIdx.x;
  if (e >= N_EDGES) return;
  int s = (int)ei[e];
  int d = (int)ei[(long long)N_EDGES + e];
  const float* xs = x + (long long)s * FEAT;
  float* ad = agg + (long long)d * FEAT;
#pragma unroll
  for (int f = 0; f < FEAT; ++f) atomicAdd(&ad[f], xs[f]);
  atomicAdd(&cnt[d], 1.0f);
}

__global__ __launch_bounds__(256) void node_gemm_kernel(
    const float* __restrict__ x, const float* __restrict__ agg,
    const float* __restrict__ cnt,
    const float* __restrict__ w_self, const float* __restrict__ b_self,
    const float* __restrict__ w_neigh, const float* __restrict__ b_neigh,
    float* __restrict__ h_out, float* __restrict__ gsum) {
  // Wt[n][k]: combined weight, transposed, f16.  k<10: w_self, 10<=k<20: w_neigh, else 0
  __shared__ _Float16 Wt[32 * 32];
  __shared__ float biasL[32];
  __shared__ float colAcc[32];
  const int tid = threadIdx.x;

  for (int i = tid; i < 32 * 32; i += 256) {
    int n = i >> 5, k = i & 31;
    float w = 0.0f;
    if (k < FEAT)          w = w_self[k * HID + n];
    else if (k < 2 * FEAT) w = w_neigh[(k - FEAT) * HID + n];
    Wt[n * 32 + k] = (_Float16)w;
  }
  if (tid < 32) {
    biasL[tid]  = b_self[tid] + b_neigh[tid];
    colAcc[tid] = 0.0f;
  }
  __syncthreads();

  const int wave  = tid >> 5;
  const int lane  = tid & 31;
  const int khalf = lane >> 4;      // which K-half of the wave this lane sits in
  const int mrow  = lane & 15;
  const int n0 = mrow, n1 = mrow + 16;

  const long long tile = (long long)blockIdx.x * 8 + wave;   // 16 nodes per tile
  float s0 = 0.0f, s1 = 0.0f;

  if (tile < (long long)(N_NODES / 16)) {   // wave-uniform guard: EXEC stays all-ones
    const long long base = tile * 16;
    const long long node = base + mrow;

    // K-vector for this node: [x(10) | agg/cnt (10) | 0-pad(12)]
    float kv[2 * FEAT];
    const float* xr = x + node * FEAT;
    const float* ar = agg + node * FEAT;
    const float inv = 1.0f / fmaxf(cnt[node], 1.0f);
#pragma unroll
    for (int f = 0; f < FEAT; ++f) { kv[f] = xr[f]; kv[FEAT + f] = ar[f] * inv; }

    // A contiguous K layout (matches B): lane<16 -> K {0..15}; lane>=16 -> K {16..31}
    v16h a;
#pragma unroll
    for (int e = 0; e < 16; ++e) {
      const int klo = e;                         // khalf==0 K index
      const int khi = e + 16;                    // khalf==1 K index
      const float v0 = (klo < 2 * FEAT) ? kv[klo] : 0.0f;
      const float v1 = (khi < 2 * FEAT) ? kv[khi] : 0.0f;
      a[e] = (_Float16)(khalf ? v1 : v0);
    }

    // B contiguous K layout (matches A): lanes 0..15 hold K=0..15, lanes 16..31 hold K=16..31 (contiguous in Wt)
    const v16h b0 = *(const v16h*)&Wt[n0 * 32 + khalf * 16];
    const v16h b1 = *(const v16h*)&Wt[n1 * 32 + khalf * 16];

    // C = bias broadcast (col = lane&15 for every C VGPR)
    v8f c0, c1;
    const float bias0 = biasL[n0], bias1 = biasL[n1];
#pragma unroll
    for (int r = 0; r < 8; ++r) { c0[r] = bias0; c1[r] = bias1; }

    c0 = __builtin_amdgcn_wmma_f32_16x16x32_f16(false, a, false, b0, (short)0, c0, false, false);
    c1 = __builtin_amdgcn_wmma_f32_16x16x32_f16(false, a, false, b1, (short)0, c1, false, false);

    // D layout: VGPR r -> row M = r + khalf*8, col N = lane&15 (+16 for tile 1)
    const long long rowbase = base + khalf * 8;
#pragma unroll
    for (int r = 0; r < 8; ++r) {
      const float v0 = fmaxf(c0[r], 0.0f);
      const float v1 = fmaxf(c1[r], 0.0f);
      h_out[(rowbase + r) * HID + n0] = v0;
      h_out[(rowbase + r) * HID + n1] = v1;
      s0 += v0; s1 += v1;
    }
  }

  // block-level column reduction for graph embedding
  atomicAdd(&colAcc[n0], s0);
  atomicAdd(&colAcc[n1], s1);
  __syncthreads();
  if (tid < 32) atomicAdd(&gsum[tid], colAcc[tid]);
}

__global__ void finalize_kernel(const float* __restrict__ gsum,
                                const float* __restrict__ w_act, const float* __restrict__ b_act,
                                const float* __restrict__ w_prim, const float* __restrict__ b_prim,
                                float* __restrict__ out) {
  __shared__ float ge[32];
  const int tid = threadIdx.x;
  if (tid < 32) {
    const float g = gsum[tid] * (1.0f / (float)N_NODES);
    ge[tid] = g;
    out[N_TOOLS + N_PRIM + tid] = g;   // graph_embed at offset 21
  }
  __syncthreads();
  if (tid < N_TOOLS) {
    float acc = b_act[tid];
    for (int k = 0; k < HID; ++k) acc += ge[k] * w_act[k * N_TOOLS + tid];
    out[tid] = acc;                    // action_logits at offset 0
  } else if (tid >= 32 && tid < 32 + N_PRIM) {
    const int t = tid - 32;
    float acc = b_prim[t];
    for (int k = 0; k < HID; ++k) acc += ge[k] * w_prim[k * N_PRIM + t];
    out[N_TOOLS + t] = acc;            // primitive_logits at offset 13
  }
}

extern "C" void kernel_launch(void* const* d_in, const int* in_sizes, int n_in,
                              void* d_out, int out_size, void* d_ws, size_t ws_size,
                              hipStream_t stream) {
  (void)in_sizes; (void)n_in; (void)out_size; (void)ws_size;
  const float*     x       = (const float*)d_in[0];
  const long long* ei      = (const long long*)d_in[1];
  const float*     w_self  = (const float*)d_in[2];
  const float*     b_self  = (const float*)d_in[3];
  const float*     w_neigh = (const float*)d_in[4];
  const float*     b_neigh = (const float*)d_in[5];
  const float*     w_act   = (const float*)d_in[6];
  const float*     b_act   = (const float*)d_in[7];
  const float*     w_prim  = (const float*)d_in[8];
  const float*     b_prim  = (const float*)d_in[9];

  float* out  = (float*)d_out;
  float* ws   = (float*)d_ws;
  float* agg  = ws;
  float* cnt  = ws + (size_t)N_NODES * FEAT;
  float* gsum = cnt + N_NODES;
  float* h_out = out + N_TOOLS + N_PRIM + HID;   // h region at offset 53

  const int zn = N_NODES * FEAT + N_NODES + 32;
  zero_ws_kernel<<<2048, 256, 0, stream>>>(ws, zn);

  edge_scatter_kernel<<<(N_EDGES + 255) / 256, 256, 0, stream>>>(x, ei, agg, cnt);

  const int ntiles  = N_NODES / 16;         // 31250 (exact)
  const int nblocks = (ntiles + 7) / 8;     // 8 waves (tiles) per 256-thread block
  node_gemm_kernel<<<nblocks, 256, 0, stream>>>(x, agg, cnt, w_self, b_self,
                                                w_neigh, b_neigh, h_out, gsum);

  finalize_kernel<<<1, 64, 0, stream>>>(gsum, w_act, b_act, w_prim, b_prim, out);
}
